// DenseContrastiveLossV2_21423296873244
// MI455X (gfx1250) — compile-verified, matrix operand order checked by reading, not run on hardware
//
#include <hip/hip_runtime.h>
#include <hip/hip_bf16.h>

// Problem constants (from reference)
#define M_ROWS 8192   // T*V
#define CCH    256    // C
#define HWSZ   9216   // H*W
// TEMPERATURE = 0.5 -> multiply dot by 2.0
//
// Math note: log_prob = (s-m) - log(exp(s-m) + neg_sum(m)) is invariant to the
// row max m (it cancels exactly). Inputs are L2-normalized so s in [-2,2] and
// exp(s) in [0.135, 7.4]: perfectly stable in f32 with m = 0. The max pass is
// therefore eliminated; stats computes only neg_sum = sum_neg exp(s).
//
// Reduction note: neg_sum and pos_sum are linear, so each lane accumulates its
// own columns' partials in registers and the cross-lane (16-lane) reduction is
// done ONCE at kernel end, not per group.

typedef __attribute__((ext_vector_type(16))) __bf16 v16bf;
typedef __attribute__((ext_vector_type(8)))  __bf16 v8bf;
typedef __attribute__((ext_vector_type(8)))  float  v8f;

union frag16 { v16bf v; v8bf h[2]; };

__device__ __forceinline__ unsigned short f2bf(float f) {
  unsigned u = __float_as_uint(f);
  u += 0x7FFFu + ((u >> 16) & 1u);   // round-to-nearest-even
  return (unsigned short)(u >> 16);
}

// ds_swizzle_b32 XOR reductions (group-of-32 mode: imm = xor<<10 | 0x1f).
#define SWZ(x, imm) \
  __int_as_float(__builtin_amdgcn_ds_swizzle(__float_as_int(x), (imm)))

__device__ __forceinline__ float redsum16(float x) {  // sum within 16-lane half
  x += SWZ(x, 0x041f);  // xor 1
  x += SWZ(x, 0x081f);  // xor 2
  x += SWZ(x, 0x101f);  // xor 4
  x += SWZ(x, 0x201f);  // xor 8
  return x;
}
__device__ __forceinline__ float redsum32(float x) {  // sum across wave32
  x = redsum16(x);
  x += SWZ(x, 0x401f);  // xor 16
  return x;
}

// Load one 16-bit A/B fragment (ISA 7.12.2 layout) from a row-major bf16
// matrix with stride CCH. lane: idx = lane&15 (M or N), half = lane>>4.
// Elements 0..7  -> K = kbase + 8*half .. +7
// Elements 8..15 -> K = kbase + 16 + 8*half .. +7
__device__ __forceinline__ v16bf load_frag(const unsigned short* base, int idx,
                                           int kbase, int half) {
  frag16 f;
  const unsigned short* p = base + idx * CCH + kbase + 8 * half;
  f.h[0] = *(const v8bf*)(p);
  f.h[1] = *(const v8bf*)(p + 16);
  return f.v;
}

#define WMMA_BF16(A, B, C)                                                    \
  __builtin_amdgcn_wmma_f32_16x16x32_bf16(false, (A), false, (B), (short)0,   \
                                          (C), false, false)

// ---------------------------------------------------------------------------
// Kernel 1: gather sampled pixels, L2-normalize along C, emit bf16 feats.
// One block per row (T*V rows), 256 threads = one channel each.
// ---------------------------------------------------------------------------
__global__ void gather_norm_kernel(const float* __restrict__ features,
                                   const int* __restrict__ batch_inds,
                                   const long long* __restrict__ sample_inds,
                                   unsigned short* __restrict__ feats) {
  int row = blockIdx.x;
  int t = row >> 8;
  int v = row & 255;
  int c = threadIdx.x;
  long long b = batch_inds[t];
  long long s = sample_inds[t * 256 + v];
  float x = features[(b * CCH + c) * (long long)HWSZ + s];
  float x2 = redsum32(x * x);
  __shared__ float red[8];
  int wid = threadIdx.x >> 5, lane = threadIdx.x & 31;
  if (lane == 0) red[wid] = x2;
  __syncthreads();
  float tot = 0.f;
#pragma unroll
  for (int i = 0; i < 8; ++i) tot += red[i];
  float norm = fmaxf(sqrtf(tot), 1e-12f);
  feats[row * CCH + c] = f2bf(x / norm);
}

// ---------------------------------------------------------------------------
// Kernel 2: per-row neg_sum = sum over different-label cols of exp(2*dot).
// Block = 256 threads = 8 waves; each wave owns a 16-row strip (128 rows/WG),
// A fragments resident in VGPRs. Per iteration a 32-row x 256-K bf16 column
// panel (16 KB) is brought into LDS with ASYNC loads (double-buffered,
// ASYNCcnt-tracked); negative-label groups run two independent WMMA chains.
// Positive groups are skipped here (the pos kernel covers them), so each
// Gram tile is computed exactly once across the two kernels. Lanes keep
// per-lane partial sums; one cross-lane reduction at the end.
// ---------------------------------------------------------------------------
__global__ void __launch_bounds__(256)
stats_kernel(const unsigned short* __restrict__ feats,
             const int* __restrict__ labels,
             float* __restrict__ row_ns) {
  __shared__ unsigned short panel[2][32 * CCH];  // 2 x 16 KB
  int tid = threadIdx.x;
  int wid = tid >> 5, lane = tid & 31;
  int half = lane >> 4;
  int rw = blockIdx.x * 128 + wid * 16;
  int lab_r = labels[rw >> 8];   // uniform across the whole block

  v16bf afrag[8];
#pragma unroll
  for (int kk = 0; kk < 8; ++kk)
    afrag[kk] = load_frag(feats, rw + (lane & 15), kk * 32, half);

  float ns[8];
#pragma unroll
  for (int r = 0; r < 8; ++r) ns[r] = 0.f;

  // Issue async global->LDS copy of one 16 KB column panel (group = 32 rows).
  auto issue_stage = [&](int buf, int grp) {
    const char* src = (const char*)(feats + (size_t)grp * 32 * CCH);
    unsigned ldsbase = (unsigned)(size_t)&panel[buf][0];
#pragma unroll
    for (int k = 0; k < 4; ++k) {
      unsigned off = (unsigned)(tid + k * 256) * 16u;
      unsigned ldsaddr = ldsbase + off;
      const void* gaddr = src + off;
      asm volatile("global_load_async_to_lds_b128 %0, %1, off"
                   :: "v"(ldsaddr), "v"(gaddr) : "memory");
    }
  };

  issue_stage(0, 0);
  asm volatile("s_wait_asynccnt 0x0" ::: "memory");
  __syncthreads();

  for (int grp = 0; grp < M_ROWS / 32; ++grp) {
    int cur = grp & 1;
    if (grp + 1 < M_ROWS / 32) issue_stage(cur ^ 1, grp + 1);

    int lab_c = labels[grp >> 3];   // 32-col group never crosses an anchor
    if (lab_c != lab_r) {           // block-uniform: only negative groups
      const unsigned short* pan = &panel[cur][0];
      v8f acc0 = {}, acc1 = {};
#pragma unroll
      for (int kk = 0; kk < 8; ++kk) {
        v16bf b0 = load_frag(pan, (lane & 15), kk * 32, half);
        v16bf b1 = load_frag(pan, 16 + (lane & 15), kk * 32, half);
        acc0 = WMMA_BF16(afrag[kk], b0, acc0);   // two independent chains
        acc1 = WMMA_BF16(afrag[kk], b1, acc1);
      }
#pragma unroll
      for (int r = 0; r < 8; ++r)   // per-lane partial only: no cross-lane ops
        ns[r] += __expf(acc0[r] * 2.0f) + __expf(acc1[r] * 2.0f);
    }
    asm volatile("s_wait_asynccnt 0x0" ::: "memory");
    __syncthreads();
  }
  // single cross-lane reduction per row at the end
#pragma unroll
  for (int r = 0; r < 8; ++r) ns[r] = redsum16(ns[r]);
  if ((lane & 15) == 0) {
#pragma unroll
    for (int r = 0; r < 8; ++r)
      row_ns[rw + r + 8 * half] = ns[r];
  }
}

// ---------------------------------------------------------------------------
// Kernel 3: positive-pair pass. Only same-label 32-col groups are staged and
// computed (label is group-uniform and block-uniform). Accumulates
//   sum over positives of  s - log(exp(s) + neg_sum),  diag excluded,
// per lane; reduces across lanes once at the end; divides by the
// (label-only) positive count.
// ---------------------------------------------------------------------------
__global__ void __launch_bounds__(256)
pos_kernel(const unsigned short* __restrict__ feats,
           const int* __restrict__ labels,
           const float* __restrict__ row_ns,
           float* __restrict__ row_r) {
  __shared__ unsigned short panel[32 * CCH];  // 16 KB
  int tid = threadIdx.x;
  int wid = tid >> 5, lane = tid & 31;
  int half = lane >> 4;
  int rw = blockIdx.x * 128 + wid * 16;
  int lab_r = labels[rw >> 8];   // uniform across the whole block

  v16bf afrag[8];
#pragma unroll
  for (int kk = 0; kk < 8; ++kk)
    afrag[kk] = load_frag(feats, rw + (lane & 15), kk * 32, half);

  float ns[8], ps[8];
#pragma unroll
  for (int r = 0; r < 8; ++r) {
    ns[r] = row_ns[rw + r + 8 * half];
    ps[r] = 0.f;
  }
  int pgroups = 0;

  for (int grp = 0; grp < M_ROWS / 32; ++grp) {
    int lab_c = labels[grp >> 3];
    if (lab_c != lab_r) continue;  // block-uniform: barriers stay uniform
    ++pgroups;
    int cb = grp * 32;
    {
      const uint4* src = (const uint4*)(feats + (size_t)cb * CCH);
      uint4* dst = (uint4*)panel;
#pragma unroll
      for (int k = 0; k < 4; ++k) dst[tid + k * 256] = src[tid + k * 256];
    }
    __syncthreads();
    v8f acc0 = {}, acc1 = {};
#pragma unroll
    for (int kk = 0; kk < 8; ++kk) {
      v16bf b0 = load_frag(panel, (lane & 15), kk * 32, half);
      v16bf b1 = load_frag(panel, 16 + (lane & 15), kk * 32, half);
      acc0 = WMMA_BF16(afrag[kk], b0, acc0);
      acc1 = WMMA_BF16(afrag[kk], b1, acc1);
    }
#pragma unroll
    for (int r = 0; r < 8; ++r) {
      int rowg = rw + r + 8 * half;
      float s0 = acc0[r] * 2.0f;
      float s1 = acc1[r] * 2.0f;
      float v0 = s0 - __logf(__expf(s0) + ns[r]);
      float v1 = s1 - __logf(__expf(s1) + ns[r]);
      if (cb + (lane & 15) == rowg)      v0 = 0.f;  // mask diagonal
      if (cb + 16 + (lane & 15) == rowg) v1 = 0.f;
      ps[r] += v0 + v1;   // per-lane partial only
    }
    __syncthreads();
  }
  // single cross-lane reduction per row, then normalize by positive count
  float denom = fmaxf(32.f * (float)pgroups - 1.f, 1e-10f);
#pragma unroll
  for (int r = 0; r < 8; ++r) ps[r] = redsum16(ps[r]);
  if ((lane & 15) == 0) {
#pragma unroll
    for (int r = 0; r < 8; ++r)
      row_r[rw + r + 8 * half] = ps[r] / denom;
  }
}

// ---------------------------------------------------------------------------
// Kernel 4: final scalar reduction: loss = -mean(row_r)
// ---------------------------------------------------------------------------
__global__ void reduce_kernel(const float* __restrict__ row_r,
                              float* __restrict__ out) {
  float s = 0.f;
  for (int i = threadIdx.x; i < M_ROWS; i += 256) s += row_r[i];
  s = redsum32(s);
  __shared__ float red[8];
  int wid = threadIdx.x >> 5, lane = threadIdx.x & 31;
  if (lane == 0) red[wid] = s;
  __syncthreads();
  if (threadIdx.x == 0) {
    float t = 0.f;
#pragma unroll
    for (int i = 0; i < 8; ++i) t += red[i];
    out[0] = -t / (float)M_ROWS;
  }
}

extern "C" void kernel_launch(void* const* d_in, const int* in_sizes, int n_in,
                              void* d_out, int out_size, void* d_ws, size_t ws_size,
                              hipStream_t stream) {
  const float*     features    = (const float*)d_in[0];
  const int*       batch_inds  = (const int*)d_in[1];
  const long long* sample_inds = (const long long*)d_in[2];  // int64 in reference
  const int*       labels      = (const int*)d_in[3];
  float*           out         = (float*)d_out;

  // workspace layout
  unsigned short* feats = (unsigned short*)d_ws;                 // 8192*256*2 = 4 MB
  float* row_ns = (float*)((char*)d_ws + (size_t)M_ROWS * CCH * 2);
  float* row_r  = row_ns + M_ROWS;

  gather_norm_kernel<<<M_ROWS, 256, 0, stream>>>(features, batch_inds,
                                                 sample_inds, feats);
  stats_kernel<<<M_ROWS / 128, 256, 0, stream>>>(feats, labels, row_ns);
  pos_kernel<<<M_ROWS / 128, 256, 0, stream>>>(feats, labels, row_ns, row_r);
  reduce_kernel<<<1, 256, 0, stream>>>(row_r, out);
}
